// Upsampling_55559696941158
// MI455X (gfx1250) — matrix operand order repaired and checked by reading, not compiled, not run on hardware
//
#include <hip/hip_runtime.h>

// ---------------------------------------------------------------------------
// Types for CDNA5 WMMA (gfx1250, wave32)
// ---------------------------------------------------------------------------
typedef __attribute__((ext_vector_type(16))) __bf16 v16bf;
typedef __attribute__((ext_vector_type(8)))  float  v8f;

union FragU { v16bf v; uint4 q[2]; };

__device__ __forceinline__ unsigned short f2bf(float x) {
    unsigned u = __float_as_uint(x);
    unsigned r = u + 0x7FFFu + ((u >> 16) & 1u);   // round-to-nearest-even
    return (unsigned short)(r >> 16);
}

// Packed bf16 atomic add: GLOBAL_ATOMIC_PK_ADD_BF16 (CDNA5), no return.
__device__ __forceinline__ void atomic_pk_add_bf16(unsigned short* addr, unsigned data) {
    asm volatile("global_atomic_pk_add_bf16 %0, %1, off"
                 :: "v"(addr), "v"(data) : "memory");
}

// Async global -> LDS copy, 16 bytes per lane (ASYNCcnt-tracked, no VGPR data).
__device__ __forceinline__ void async_g2l_b128(unsigned lds_off, const void* gaddr) {
    asm volatile("global_load_async_to_lds_b128 %0, %1, off"
                 :: "v"(lds_off), "v"(gaddr) : "memory");
}

// ---------------------------------------------------------------------------
// Problem constants
// ---------------------------------------------------------------------------
#define NN    8192
#define EE    131072
#define NYY   16384
#define CC    64
#define KTOT  125
#define KDIM  8000          // KTOT * CC
#define NT    (KDIM / 32)   // 250 K-steps

// Dynamic-LDS layout (bytes): double-buffered {A tile 8 KB, B tile 4 KB}
#define LDS_BUF   12288u
#define LDS_A(b)  ((b) * LDS_BUF)
#define LDS_B(b)  ((b) * LDS_BUF + 8192u)
#define LDS_TOTAL 24576u

// ---------------------------------------------------------------------------
// Zero scratch (float4 grid-stride)
// ---------------------------------------------------------------------------
__global__ void zero_kernel(float* __restrict__ p, long n4) {
    long i = (long)blockIdx.x * blockDim.x + threadIdx.x;
    long stride = (long)gridDim.x * blockDim.x;
    float4 z; z.x = 0.f; z.y = 0.f; z.z = 0.f; z.w = 0.f;
    for (; i < n4; i += stride) ((float4*)p)[i] = z;
}

// ---------------------------------------------------------------------------
// One-time weight conversion: Wt[o][k] = bf16(W[k][o])   (transposed, bf16)
// ---------------------------------------------------------------------------
__global__ void convertW_kernel(const float* __restrict__ W,
                                unsigned short* __restrict__ Wt, int Kd) {
    int idx = blockIdx.x * blockDim.x + threadIdx.x;
    if (idx >= Kd * CC) return;
    int k = idx >> 6, o = idx & 63;             // coalesced read over o
    Wt[(long)o * Kd + k] = f2bf(W[idx]);
}

// ---------------------------------------------------------------------------
// Spline-basis + scatter: one wave per edge, 2 adjacent channels per lane,
// one packed-bf16 atomic per (corner, lane).
// ---------------------------------------------------------------------------
__global__ void scatter_kernel(const float* __restrict__ xin,
                               const int*   __restrict__ ei,
                               const float* __restrict__ eattr,
                               unsigned short* __restrict__ A,
                               float* __restrict__ deg,
                               int E, int addDeg)
{
    int wave = (blockIdx.x * blockDim.x + threadIdx.x) >> 5;
    int lane = threadIdx.x & 31;
    if (wave >= E) return;

    int src = ei[wave];
    int dst = ei[E + wave];

    float f[3]; int bi[3];
#pragma unroll
    for (int d = 0; d < 3; ++d) {
        float v = eattr[wave * 3 + d] * 4.0f;   // (K-1) = 4
        float b = floorf(v);
        f[d]  = v - b;
        bi[d] = (int)b;
    }

    float x0 = xin[(long)src * CC + 2 * lane];
    float x1 = xin[(long)src * CC + 2 * lane + 1];

#pragma unroll
    for (int c = 0; c < 8; ++c) {
        float bs = 1.0f;
        int wi = 0, p = 1;
#pragma unroll
        for (int d = 0; d < 3; ++d) {
            int bit = (c >> d) & 1;
            bs *= bit ? f[d] : (1.0f - f[d]);
            int t = bi[d] + bit;
            t = min(max(t, 0), 4);
            wi += t * p;
            p *= 5;
        }
        unsigned pk = ((unsigned)f2bf(bs * x1) << 16) | (unsigned)f2bf(bs * x0);
        atomic_pk_add_bf16(A + ((long)dst * KTOT + wi) * CC + 2 * lane, pk);
    }
    if (addDeg && lane == 0) atomicAdd(&deg[dst], 1.0f);
}

// Fragment builds as explicit uint4 LDS reads -> ds_load_b128.
__device__ __forceinline__ void wmma_step(const char* __restrict__ aBase,
                                          const char* __restrict__ bBase,
                                          int lane, int wave, v8f c[4])
{
    int half = lane >> 4;
    int m    = lane & 15;

    // A fragment 16x32 bf16: row m of the wave's 16-row strip; each row is
    // 64 B = 4 uint4; quads {half, 2+half} per ISA layout.
    const uint4* rowA = (const uint4*)__builtin_assume_aligned(
        aBase + (wave * 16 + m) * 64, 16);
    FragU a;
    a.q[0] = rowA[half];
    a.q[1] = rowA[2 + half];

#pragma unroll
    for (int nt = 0; nt < 4; ++nt) {
        // B fragment 32x16 (N-major rows of 64 B): quads {half, 2+half}
        const uint4* rowB = (const uint4*)__builtin_assume_aligned(
            bBase + (nt * 16 + m) * 64, 16);
        FragU b;
        b.q[0] = rowB[half];
        b.q[1] = rowB[2 + half];

        c[nt] = __builtin_amdgcn_wmma_f32_16x16x32_bf16(
            false, a.v, false, b.v, (short)0, c[nt], false, false);
    }
}

// ---------------------------------------------------------------------------
// Fused SplineConv apply: out = ELU( (A @ W)/deg + x @ Wroot + bias )
// Block: 128 M-rows x 64 N-cols, 8 waves; double-buffered async-LDS K loop
// with a single (non-unrolled) body and peeled epilogue.
// ---------------------------------------------------------------------------
__global__ void __launch_bounds__(256)
gemm_kernel(const unsigned short* __restrict__ A,      // [N][8000] bf16
            const unsigned short* __restrict__ Wt,     // [64][8000] bf16 (transposed)
            const float* __restrict__ xin,             // [N][64] f32
            const unsigned short* __restrict__ Wroott, // [64][64] bf16 (transposed)
            const float* __restrict__ bias,            // [64]
            const float* __restrict__ deg,             // [N]
            float* __restrict__ out)                   // [N][64]
{
    extern __shared__ __align__(16) char smem[];       // 24 KB, dynamic => offset 0

    int tid  = threadIdx.x;
    int lane = tid & 31;
    int wave = tid >> 5;
    int rowBase = blockIdx.x * 128;
    int half = lane >> 4;

    v8f c[4];
#pragma unroll
    for (int nt = 0; nt < 4; ++nt)
#pragma unroll
        for (int r = 0; r < 8; ++r) c[nt][r] = 0.0f;

    // Per-thread async-copy source pointers / LDS offsets (3 b128 per tile).
    int rq = tid >> 2, q8 = (tid & 3) * 8;             // A rows 0..63 / 64..127; B rows 0..63
    const unsigned short* gA0 = A + (long)(rowBase + rq) * KDIM + q8;
    const unsigned short* gA1 = A + (long)(rowBase + 64 + rq) * KDIM + q8;
    const unsigned short* gB  = Wt + (long)rq * KDIM + q8;
    unsigned oA0 = (unsigned)(rq * 32 + q8) * 2;
    unsigned oA1 = (unsigned)((64 + rq) * 32 + q8) * 2;
    unsigned oB  = 8192u + (unsigned)(rq * 32 + q8) * 2;

    // ---- prologue: tile 0 -> buf 0 ----
    async_g2l_b128(oA0, gA0);
    async_g2l_b128(oA1, gA1);
    async_g2l_b128(oB,  gB);
    gA0 += 32; gA1 += 32; gB += 32;

    // ---- single-body double-buffered K loop (tiles 0..NT-2) ----
    unsigned cur = 0;
#pragma clang loop unroll(disable)
    for (int kt = 0; kt < NT - 1; ++kt) {
        unsigned nb = (cur ^ 1u) * LDS_BUF;            // prefetch into other buffer
        async_g2l_b128(oA0 + nb, gA0);
        async_g2l_b128(oA1 + nb, gA1);
        async_g2l_b128(oB  + nb, gB);
        gA0 += 32; gA1 += 32; gB += 32;
        asm volatile("s_wait_asynccnt 3" ::: "memory");    // tile kt landed
        __syncthreads();
        wmma_step(smem + cur * LDS_BUF, smem + cur * LDS_BUF + 8192u, lane, wave, c);
        __syncthreads();                               // reads done before reuse
        cur ^= 1u;
    }

    // ---- epilogue: last tile ----
    asm volatile("s_wait_asynccnt 0" ::: "memory");
    __syncthreads();
    wmma_step(smem + cur * LDS_BUF, smem + cur * LDS_BUF + 8192u, lane, wave, c);

    // ---- mean aggregation: multiply conv part by 1/deg (v_rcp_f32) ----
    float rdg[8];
#pragma unroll
    for (int r = 0; r < 8; ++r)
        rdg[r] = __builtin_amdgcn_rcpf(
            fmaxf(deg[rowBase + wave * 16 + r + 8 * half], 1.0f));
#pragma unroll
    for (int nt = 0; nt < 4; ++nt)
#pragma unroll
        for (int r = 0; r < 8; ++r) c[nt][r] = c[nt][r] * rdg[r];

    // ---- root weight: + x @ Wroot (two more K=32 WMMA steps, VGPR staging) ----
    unsigned short (*lA0)[32] = (unsigned short (*)[32])(smem + LDS_A(0));
    unsigned short (*lB0)[32] = (unsigned short (*)[32])(smem + LDS_B(0));
    for (int kt = 0; kt < 2; ++kt) {
        __syncthreads();
        // stage A part from f32 xin with conversion
#pragma unroll
        for (int j = 0; j < 4; ++j) {
            int idx = tid + j * 256;
            int row = idx >> 3, q = idx & 7;
            float4 v = *(const float4*)(xin + (long)(rowBase + row) * CC + kt * 32 + q * 4);
            lA0[row][q * 4 + 0] = f2bf(v.x);
            lA0[row][q * 4 + 1] = f2bf(v.y);
            lA0[row][q * 4 + 2] = f2bf(v.z);
            lA0[row][q * 4 + 3] = f2bf(v.w);
        }
        // stage B part from pre-converted transposed Wroot
        {
            int o = tid >> 2, q = tid & 3;
            *(uint4*)&lB0[o][q * 8] =
                *(const uint4*)(Wroott + (long)o * CC + kt * 32 + q * 8);
        }
        __syncthreads();
        wmma_step((const char*)lA0, (const char*)lB0, lane, wave, c);
    }

    // ---- bias + ELU + store ----
    int col0 = lane & 15;
#pragma unroll
    for (int nt = 0; nt < 4; ++nt) {
        int col = nt * 16 + col0;
        float bv = bias[col];
#pragma unroll
        for (int r = 0; r < 8; ++r) {
            int row = rowBase + wave * 16 + r + 8 * half;
            float v = c[nt][r] + bv;
            v = (v > 0.0f) ? v : (__expf(v) - 1.0f);
            out[(long)row * CC + col] = v;
        }
    }
}

// ---------------------------------------------------------------------------
// knn_interpolate (k = 3): brute force, one thread per output point.
// ---------------------------------------------------------------------------
__global__ void knn_kernel(const float* __restrict__ h,
                           const float* __restrict__ posx,
                           const float* __restrict__ posy,
                           const int*   __restrict__ bx,
                           const int*   __restrict__ by,
                           float* __restrict__ out,
                           int Nx, int Ny)
{
    int y = blockIdx.x * blockDim.x + threadIdx.x;
    if (y >= Ny) return;

    float off = 1e5f * (float)by[y];
    float p0 = posy[y * 3 + 0] + off;
    float p1 = posy[y * 3 + 1] + off;
    float p2 = posy[y * 3 + 2] + off;

    float d0 = 3.4e38f, d1 = 3.4e38f, d2 = 3.4e38f;
    int   i0 = 0, i1 = 0, i2 = 0;

    for (int i = 0; i < Nx; ++i) {
        float xo = 1e5f * (float)bx[i];
        float dx = posx[i * 3 + 0] + xo - p0;
        float dy = posx[i * 3 + 1] + xo - p1;
        float dz = posx[i * 3 + 2] + xo - p2;
        float d  = dx * dx + dy * dy + dz * dz;
        if (d < d2) {
            if (d < d0)      { d2 = d1; i2 = i1; d1 = d0; i1 = i0; d0 = d; i0 = i; }
            else if (d < d1) { d2 = d1; i2 = i1; d1 = d; i1 = i; }
            else             { d2 = d; i2 = i; }
        }
    }

    float w0 = 1.0f / fmaxf(d0, 1e-16f);
    float w1 = 1.0f / fmaxf(d1, 1e-16f);
    float w2 = 1.0f / fmaxf(d2, 1e-16f);
    float ws = 1.0f / (w0 + w1 + w2);

    const float* h0 = h + (long)i0 * CC;
    const float* h1 = h + (long)i1 * CC;
    const float* h2 = h + (long)i2 * CC;
    float* o = out + (long)y * CC;
#pragma unroll 8
    for (int cch = 0; cch < CC; ++cch)
        o[cch] = (w0 * h0[cch] + w1 * h1[cch] + w2 * h2[cch]) * ws;
}

// ---------------------------------------------------------------------------
// Launch
// ---------------------------------------------------------------------------
extern "C" void kernel_launch(void* const* d_in, const int* in_sizes, int n_in,
                              void* d_out, int out_size, void* d_ws, size_t ws_size,
                              hipStream_t stream)
{
    const float* x     = (const float*)d_in[0];
    const int*   ei    = (const int*)  d_in[1];
    const float* ea    = (const float*)d_in[2];
    const float* posx  = (const float*)d_in[3];
    const int*   bx    = (const int*)  d_in[4];
    const float* posy  = (const float*)d_in[5];
    const int*   by    = (const int*)  d_in[6];
    const float* Wa    = (const float*)d_in[7];
    const float* roota = (const float*)d_in[8];
    const float* biasa = (const float*)d_in[9];
    const float* Wb    = (const float*)d_in[10];
    const float* rootb = (const float*)d_in[11];
    const float* biasb = (const float*)d_in[12];
    float* out = (float*)d_out;

    // Workspace: A(bf16) | deg | h1 | h2 | Wta | Wtb | Wroota_t | Wrootb_t
    unsigned short* A   = (unsigned short*)d_ws;            // NN*KDIM bf16
    float* deg = (float*)(A + (long)NN * KDIM);             // NN
    float* h1  = deg + NN;
    float* h2  = h1 + (long)NN * CC;
    unsigned short* Wta = (unsigned short*)(h2 + (long)NN * CC);
    unsigned short* Wtb = Wta + (long)CC * KDIM;
    unsigned short* Wra = Wtb + (long)CC * KDIM;
    unsigned short* Wrb = Wra + CC * CC;

    long zAdeg = ((long)NN * KDIM * 2 + NN * 4) / 16;       // A + deg, in float4
    long zA    = ((long)NN * KDIM * 2) / 16;                // A only

    // One-time weight conversion (bf16, transposed)
    convertW_kernel<<<(KDIM * CC + 255) / 256, 256, 0, stream>>>(Wa, Wta, KDIM);
    convertW_kernel<<<(KDIM * CC + 255) / 256, 256, 0, stream>>>(Wb, Wtb, KDIM);
    convertW_kernel<<<(CC * CC + 255) / 256, 256, 0, stream>>>(roota, Wra, CC);
    convertW_kernel<<<(CC * CC + 255) / 256, 256, 0, stream>>>(rootb, Wrb, CC);

    // Layer 1
    zero_kernel   <<<4096, 256, 0, stream>>>((float*)A, zAdeg);
    scatter_kernel<<<EE / 8, 256, 0, stream>>>(x, ei, ea, A, deg, EE, 1);
    gemm_kernel   <<<NN / 128, 256, LDS_TOTAL, stream>>>(A, Wta, x, Wra, biasa, deg, h1);

    // Layer 2 (deg unchanged; rebuild A from h1)
    zero_kernel   <<<4096, 256, 0, stream>>>((float*)A, zA);
    scatter_kernel<<<EE / 8, 256, 0, stream>>>(h1, ei, ea, A, deg, EE, 0);
    gemm_kernel   <<<NN / 128, 256, LDS_TOTAL, stream>>>(A, Wtb, h1, Wrb, biasb, deg, h2);

    // KNN interpolation to output points
    knn_kernel<<<NYY / 256, 256, 0, stream>>>(h2, posx, posy, bx, by, out, NN, NYY);
}
